// NequIP_31885837205753
// MI455X (gfx1250) — matrix operand order repaired and checked by reading, not compiled, not run on hardware
//
#include <hip/hip_runtime.h>
#include <hip/hip_bf16.h>
#include <math.h>

typedef __attribute__((ext_vector_type(16))) _Float16 v16h;
typedef __attribute__((ext_vector_type(8)))  _Float16 v8h;
typedef __attribute__((ext_vector_type(8)))  float    v8f;

#define RC_F 5.0f
#define PI_F 3.14159265358979323846f

// silu via fast hardware reciprocal (v_rcp_f32) instead of IEEE divide chain
__device__ __forceinline__ float silu_f(float x) {
    return x * __builtin_amdgcn_rcpf(1.0f + __expf(-x));
}

// ---------------------------------------------------------------------------
// WMMA fragment helpers (gfx1250 wave32 layouts, 16x16x32 f16 -> f32)
//
// A (16x32 f16): lane L holds row m=L&15; lanes 0-15 hold K=0..15,
// lanes 16-31 hold K=16..31 (contiguous K, same order as B).
// B (32x16 f16): lane L holds column n=L&15; lanes 0-15 hold K=0..15,
// lanes 16-31 hold K=16..31, 2 f16 per VGPR.
// C/D (16x16 f32): VGPR j holds row m=j+8*(L/16), column n=L&15.
// ---------------------------------------------------------------------------
__device__ __forceinline__ v16h load_a_frag(const _Float16* As, int rbase, int lda, int lane) {
    int m  = lane & 15;
    int lo = (lane >> 4) << 4;   // 0 or 16
    const _Float16* row = As + (size_t)(rbase + m) * lda;
    v8h p0 = *(const v8h*)(row + lo);        // K = lo .. lo+7
    v8h p1 = *(const v8h*)(row + lo + 8);    // K = lo+8 .. lo+15
    return __builtin_shufflevector(p0, p1, 0,1,2,3,4,5,6,7,8,9,10,11,12,13,14,15);
}

__device__ __forceinline__ v16h load_b_frag(const _Float16* Bs, int cbase, int ldb, int lane) {
    int n  = lane & 15;
    int kb = (lane >> 4) << 4;   // 0 or 16
    v16h b;
#pragma unroll
    for (int i = 0; i < 16; ++i) b[i] = Bs[(size_t)(kb + i) * ldb + cbase + n];
    return b;
}

__device__ __forceinline__ v8f wmma_f16(v16h a, v16h b, v8f c) {
    return __builtin_amdgcn_wmma_f32_16x16x32_f16(false, a, false, b, (short)0, c, false, false);
}

// ---------------------------------------------------------------------------
// Kernels
// ---------------------------------------------------------------------------
__global__ void zero_kernel(float* __restrict__ p, long long count) {
    long long i = (long long)blockIdx.x * blockDim.x + threadIdx.x;
    long long stride = (long long)gridDim.x * blockDim.x;
    for (; i < count; i += stride) p[i] = 0.0f;
}

__global__ void init_nodes(const int* __restrict__ z, const float* __restrict__ emb0,
                           float* __restrict__ x0, float* __restrict__ x1,
                           float* __restrict__ x2, int N) {
    int n = blockIdx.x * blockDim.x + threadIdx.x;
    if (n >= N) return;
    int zi = z[n];
#pragma unroll
    for (int i = 0; i < 16; ++i) x0[(size_t)n * 16 + i] = emb0[(size_t)zi * 16 + i];
#pragma unroll
    for (int i = 0; i < 24; ++i) x1[(size_t)n * 24 + i] = 0.0f;
#pragma unroll
    for (int i = 0; i < 20; ++i) x2[(size_t)n * 20 + i] = 0.0f;
}

#define EK_BLOCK 128
#define EK_WAVES 4

__global__ __launch_bounds__(EK_BLOCK)
void edge_kernel(const int* __restrict__ ei, const float* __restrict__ pos,
                 const float* __restrict__ x0, const float* __restrict__ x1,
                 const float* __restrict__ x2,
                 float* __restrict__ a0, float* __restrict__ a1, float* __restrict__ a2,
                 const float* __restrict__ Wr1, const float* __restrict__ br1,
                 const float* __restrict__ Wr2, const float* __restrict__ br2,
                 const float* __restrict__ lin00, const float* __restrict__ lin01,
                 const float* __restrict__ lin11, const float* __restrict__ lin02,
                 const float* __restrict__ lin22, int E) {
    // f16 weight tiles (shared by all waves in block)
    __shared__ __align__(16) _Float16 Wr1p[32][32];   // K=8 real (pad 32), N=32
    __shared__ __align__(16) _Float16 Wr2p[32][48];   // K=32,            N=40 (pad 48)
    __shared__ __align__(16) _Float16 Lcat[32][32];   // K=16 real,       N=28 (pad 32)
    __shared__ float L11[64];                          // lin11 8x8
    __shared__ float L22[16];                          // lin22 4x4
    // per-wave staging
    __shared__ __align__(16) _Float16 As[EK_WAVES][32][32];  // A operand tile
    __shared__ float Rs[EK_WAVES][32][40];                   // radial outputs
    __shared__ float Ls[EK_WAVES][32][28];                   // s @ [lin00|lin01|lin02]

    const int tid = threadIdx.x;
    for (int idx = tid; idx < 32 * 32; idx += EK_BLOCK) {
        int k = idx >> 5, n = idx & 31;
        Wr1p[k][n] = (_Float16)((k < 8) ? Wr1[k * 32 + n] : 0.0f);
        float lv = 0.0f;
        if (k < 16) {
            if (n < 16)      lv = lin00[k * 16 + n];
            else if (n < 24) lv = lin01[k * 8 + (n - 16)];
            else if (n < 28) lv = lin02[k * 4 + (n - 24)];
        }
        Lcat[k][n] = (_Float16)lv;
    }
    for (int idx = tid; idx < 32 * 48; idx += EK_BLOCK) {
        int k = idx / 48, n = idx % 48;
        Wr2p[k][n] = (_Float16)((n < 40) ? Wr2[k * 40 + n] : 0.0f);
    }
    for (int idx = tid; idx < 64; idx += EK_BLOCK) L11[idx] = lin11[idx];
    for (int idx = tid; idx < 16; idx += EK_BLOCK) L22[idx] = lin22[idx];
    __syncthreads();

    const int lane = tid & 31;
    const int w    = tid >> 5;
    const int nn   = lane & 15;
    const int mo   = (lane >> 4) << 3;

    // Preload B fragments (resident across the whole loop)
    v16h bWr1[2], bWr2[3], bLc[2];
#pragma unroll
    for (int ct = 0; ct < 2; ++ct) bWr1[ct] = load_b_frag(&Wr1p[0][0], ct * 16, 32, lane);
#pragma unroll
    for (int ct = 0; ct < 3; ++ct) bWr2[ct] = load_b_frag(&Wr2p[0][0], ct * 16, 48, lane);
#pragma unroll
    for (int ct = 0; ct < 2; ++ct) bLc[ct]  = load_b_frag(&Lcat[0][0], ct * 16, 32, lane);

    const int ngroups = (E + 31) >> 5;
    for (int g = blockIdx.x * EK_WAVES + w; g < ngroups; g += gridDim.x * EK_WAVES) {
        int  e     = g * 32 + lane;
        bool valid = (e < E);
        int  ec    = valid ? e : (E - 1);
        int  src   = ei[ec];
        int  dst   = ei[(size_t)E + ec];

        float px = pos[(size_t)src * 3 + 0] - pos[(size_t)dst * 3 + 0];
        float py = pos[(size_t)src * 3 + 1] - pos[(size_t)dst * 3 + 1];
        float pz = pos[(size_t)src * 3 + 2] - pos[(size_t)dst * 3 + 2];
        float r    = sqrtf(px * px + py * py + pz * pz);
        float rinv = __builtin_amdgcn_rcpf(r + 1e-9f);
        float hx = px * rinv, hy = py * rinv, hz = pz * rinv;

        // sin((k+1)*theta) by Chebyshev recurrence: 1 sin + 1 cos total.
        float theta = (PI_F / RC_F) * r;
        float s1 = __sinf(theta);
        float c1 = __cosf(theta);
        float fc = (r < RC_F) ? 0.5f * (c1 + 1.0f) : 0.0f;   // 0.5*(cos(pi*clip(r/RC,0,1))+1)
        float twoc = 2.0f * c1;

        // --- stage radial basis (32 edges x 8, pad K to 32) ---
        __builtin_amdgcn_wave_barrier();
        {
            float skm1 = 0.0f, sk = s1;
#pragma unroll
            for (int k = 0; k < 8; ++k) {
                As[w][lane][k] = (_Float16)(sk * rinv);
                float nxt = twoc * sk - skm1;
                skm1 = sk;
                sk = nxt;
            }
        }
#pragma unroll
        for (int k = 8; k < 32; ++k) As[w][lane][k] = (_Float16)0.0f;
        __builtin_amdgcn_wave_barrier();

        // --- radial layer 1: (32x32) @ (32x32) ---
        v16h af0 = load_a_frag(&As[w][0][0], 0, 32, lane);
        v16h af1 = load_a_frag(&As[w][0][0], 16, 32, lane);
        v8f h00 = {}, h01 = {}, h10 = {}, h11 = {};
        h00 = wmma_f16(af0, bWr1[0], h00);
        h01 = wmma_f16(af0, bWr1[1], h01);
        h10 = wmma_f16(af1, bWr1[0], h10);
        h11 = wmma_f16(af1, bWr1[1], h11);
        __builtin_amdgcn_wave_barrier();
        {
            float b1a = br1[nn], b1b = br1[16 + nn];
#pragma unroll
            for (int j = 0; j < 8; ++j) {
                As[w][mo + j][nn]           = (_Float16)silu_f(h00[j] + b1a);
                As[w][mo + j][16 + nn]      = (_Float16)silu_f(h01[j] + b1b);
                As[w][16 + mo + j][nn]      = (_Float16)silu_f(h10[j] + b1a);
                As[w][16 + mo + j][16 + nn] = (_Float16)silu_f(h11[j] + b1b);
            }
        }
        __builtin_amdgcn_wave_barrier();

        // --- radial layer 2: (32x32) @ (32x40) ---
        af0 = load_a_frag(&As[w][0][0], 0, 32, lane);
        af1 = load_a_frag(&As[w][0][0], 16, 32, lane);
#pragma unroll
        for (int ct = 0; ct < 3; ++ct) {
            v8f c0 = {}, c1v = {};
            c0  = wmma_f16(af0, bWr2[ct], c0);
            c1v = wmma_f16(af1, bWr2[ct], c1v);
            int col = ct * 16 + nn;
            if (col < 40) {
                float bias = br2[col];
#pragma unroll
                for (int j = 0; j < 8; ++j) {
                    Rs[w][mo + j][col]      = c0[j] + bias;
                    Rs[w][16 + mo + j][col] = c1v[j] + bias;
                }
            }
        }
        __builtin_amdgcn_wave_barrier();

        // --- scalar projections: s @ [lin00|lin01|lin02]  (32x16 pad32 @ 32x28) ---
#pragma unroll
        for (int k = 0; k < 16; ++k) As[w][lane][k] = (_Float16)x0[(size_t)src * 16 + k];
#pragma unroll
        for (int k = 16; k < 32; ++k) As[w][lane][k] = (_Float16)0.0f;
        __builtin_amdgcn_wave_barrier();

        af0 = load_a_frag(&As[w][0][0], 0, 32, lane);
        af1 = load_a_frag(&As[w][0][0], 16, 32, lane);
#pragma unroll
        for (int ct = 0; ct < 2; ++ct) {
            v8f c0 = {}, c1v = {};
            c0  = wmma_f16(af0, bLc[ct], c0);
            c1v = wmma_f16(af1, bLc[ct], c1v);
            int col = ct * 16 + nn;
            if (col < 28) {
#pragma unroll
                for (int j = 0; j < 8; ++j) {
                    Ls[w][mo + j][col]      = c0[j];
                    Ls[w][16 + mo + j][col] = c1v[j];
                }
            }
        }
        __builtin_amdgcn_wave_barrier();

        // --- per-edge message assembly + scatter-add ---
        if (valid) {
            const float* Rrow = &Rs[w][lane][0];
            const float* Lrow = &Ls[w][lane][0];
            // m0 = R0 * (s @ lin00)
#pragma unroll
            for (int i = 0; i < 16; ++i) {
                atomicAdd(&a0[(size_t)dst * 16 + i], fc * Rrow[i] * Lrow[i]);
            }
            // m1 = R1a * (s@lin01) * Y1 + R1b * (x1[src] @ lin11)
            float x1s[24];
#pragma unroll
            for (int i = 0; i < 24; ++i) x1s[i] = x1[(size_t)src * 24 + i];
            float rh[3] = {hx, hy, hz};
#pragma unroll
            for (int k = 0; k < 8; ++k) {
                float r1a = fc * Rrow[16 + k];
                float r1b = fc * Rrow[24 + k];
                float t1  = Lrow[16 + k];
#pragma unroll
                for (int c = 0; c < 3; ++c) {
                    float w1 = 0.0f;
#pragma unroll
                    for (int j = 0; j < 8; ++j) w1 += x1s[j * 3 + c] * L11[j * 8 + k];
                    atomicAdd(&a1[(size_t)dst * 24 + k * 3 + c], r1a * t1 * rh[c] + r1b * w1);
                }
            }
            // m2 = R2a * (s@lin02) * Y2 + R2b * (x2[src] @ lin22)
            float x2s[20];
#pragma unroll
            for (int i = 0; i < 20; ++i) x2s[i] = x2[(size_t)src * 20 + i];
            float Y2[5] = {hx * hy, hy * hz, 3.0f * hz * hz - 1.0f, hx * hz, hx * hx - hy * hy};
#pragma unroll
            for (int k = 0; k < 4; ++k) {
                float r2a = fc * Rrow[32 + k];
                float r2b = fc * Rrow[36 + k];
                float t2  = Lrow[24 + k];
#pragma unroll
                for (int c = 0; c < 5; ++c) {
                    float w2 = 0.0f;
#pragma unroll
                    for (int j = 0; j < 4; ++j) w2 += x2s[j * 5 + c] * L22[j * 4 + k];
                    atomicAdd(&a2[(size_t)dst * 20 + k * 5 + c], r2a * t2 * Y2[c] + r2b * w2);
                }
            }
        }
    }
}

__global__ void node_update(float* __restrict__ x0, float* __restrict__ x1, float* __restrict__ x2,
                            const float* __restrict__ a0, const float* __restrict__ a1,
                            const float* __restrict__ a2,
                            const float* __restrict__ U0, const float* __restrict__ b0,
                            const float* __restrict__ U1, const float* __restrict__ U2, int N) {
    int n = blockIdx.x * blockDim.x + threadIdx.x;
    if (n >= N) return;
    float ar[16];
#pragma unroll
    for (int i = 0; i < 16; ++i) ar[i] = a0[(size_t)n * 16 + i];
#pragma unroll
    for (int i = 0; i < 16; ++i) {
        float acc = b0[i];
#pragma unroll
        for (int k = 0; k < 16; ++k) acc += ar[k] * U0[k * 16 + i];
        x0[(size_t)n * 16 + i] += silu_f(acc);
    }
    float a1r[24];
#pragma unroll
    for (int i = 0; i < 24; ++i) a1r[i] = a1[(size_t)n * 24 + i];
#pragma unroll
    for (int k = 0; k < 8; ++k)
#pragma unroll
        for (int c = 0; c < 3; ++c) {
            float acc = 0.0f;
#pragma unroll
            for (int j = 0; j < 8; ++j) acc += a1r[j * 3 + c] * U1[j * 8 + k];
            x1[(size_t)n * 24 + k * 3 + c] += acc;
        }
    float a2r[20];
#pragma unroll
    for (int i = 0; i < 20; ++i) a2r[i] = a2[(size_t)n * 20 + i];
#pragma unroll
    for (int k = 0; k < 4; ++k)
#pragma unroll
        for (int c = 0; c < 5; ++c) {
            float acc = 0.0f;
#pragma unroll
            for (int j = 0; j < 4; ++j) acc += a2r[j * 5 + c] * U2[j * 4 + k];
            x2[(size_t)n * 20 + k * 5 + c] += acc;
        }
}

__global__ void readout_kernel(const float* __restrict__ x0, const float* __restrict__ x1,
                               const float* __restrict__ x2,
                               const int* __restrict__ z, const int* __restrict__ batch,
                               const float* __restrict__ Wo1, const float* __restrict__ bo1,
                               const float* __restrict__ Wo2, const float* __restrict__ bo2,
                               const float* __restrict__ sbias, float* __restrict__ out, int N) {
    int n = blockIdx.x * blockDim.x + threadIdx.x;
    if (n >= N) return;
    float inv[28];
#pragma unroll
    for (int i = 0; i < 16; ++i) inv[i] = x0[(size_t)n * 16 + i];
#pragma unroll
    for (int j = 0; j < 8; ++j) {
        float s = 0.0f;
#pragma unroll
        for (int c = 0; c < 3; ++c) {
            float v = x1[(size_t)n * 24 + j * 3 + c];
            s += v * v;
        }
        inv[16 + j] = s;
    }
#pragma unroll
    for (int j = 0; j < 4; ++j) {
        float s = 0.0f;
#pragma unroll
        for (int c = 0; c < 5; ++c) {
            float v = x2[(size_t)n * 20 + j * 5 + c];
            s += v * v;
        }
        inv[24 + j] = s;
    }
    float e = bo2[0] + sbias[z[n]];
#pragma unroll
    for (int i = 0; i < 16; ++i) {
        float acc = bo1[i];
#pragma unroll
        for (int j = 0; j < 28; ++j) acc += inv[j] * Wo1[j * 16 + i];
        e += silu_f(acc) * Wo2[i];
    }
    atomicAdd(&out[batch[n]], e);
}

// ---------------------------------------------------------------------------
// Launch
// ---------------------------------------------------------------------------
extern "C" void kernel_launch(void* const* d_in, const int* in_sizes, int n_in,
                              void* d_out, int out_size, void* d_ws, size_t ws_size,
                              hipStream_t stream) {
    (void)n_in; (void)ws_size;
    const int*   z      = (const int*)d_in[0];
    const float* pos    = (const float*)d_in[1];
    const int*   batch  = (const int*)d_in[2];
    const int*   ei     = (const int*)d_in[3];
    const float* emb0   = (const float*)d_in[4];
    const float* Wr1    = (const float*)d_in[5];
    const float* br1    = (const float*)d_in[6];
    const float* Wr2    = (const float*)d_in[7];
    const float* br2    = (const float*)d_in[8];
    const float* lin00  = (const float*)d_in[9];
    const float* lin01  = (const float*)d_in[10];
    const float* lin11  = (const float*)d_in[11];
    const float* lin02  = (const float*)d_in[12];
    const float* lin22  = (const float*)d_in[13];
    const float* U0     = (const float*)d_in[14];
    const float* b0     = (const float*)d_in[15];
    const float* U1     = (const float*)d_in[16];
    const float* U2     = (const float*)d_in[17];
    const float* Wo1    = (const float*)d_in[18];
    const float* bo1    = (const float*)d_in[19];
    const float* Wo2    = (const float*)d_in[20];
    const float* bo2    = (const float*)d_in[21];
    const float* sbias  = (const float*)d_in[22];

    const int N = in_sizes[0];
    const int E = in_sizes[3] / 2;

    float* ws = (float*)d_ws;
    float* x0 = ws;                       // N*16
    float* x1 = x0 + (size_t)N * 16;      // N*24
    float* x2 = x1 + (size_t)N * 24;      // N*20
    float* a0 = x2 + (size_t)N * 20;      // N*16
    float* a1 = a0 + (size_t)N * 16;      // N*24
    float* a2 = a1 + (size_t)N * 24;      // N*20

    const int nblocks = (N + 255) / 256;
    init_nodes<<<nblocks, 256, 0, stream>>>(z, emb0, x0, x1, x2, N);

    const int ngroups = (E + 31) / 32;
    int eblocks = (ngroups + EK_WAVES - 1) / EK_WAVES;
    if (eblocks > 2048) eblocks = 2048;   // ~6 tiles per wave: amortize weight staging
    if (eblocks < 1) eblocks = 1;

    const long long acount = (long long)N * 60;
    int zblocks = (int)((acount + 255) / 256);
    if (zblocks > 8192) zblocks = 8192;

    for (int layer = 0; layer < 2; ++layer) {
        zero_kernel<<<zblocks, 256, 0, stream>>>(a0, acount);
        edge_kernel<<<eblocks, EK_BLOCK, 0, stream>>>(
            ei, pos, x0, x1, x2, a0, a1, a2,
            Wr1, br1, Wr2, br2, lin00, lin01, lin11, lin02, lin22, E);
        node_update<<<nblocks, 256, 0, stream>>>(x0, x1, x2, a0, a1, a2, U0, b0, U1, U2, N);
    }

    zero_kernel<<<1, 256, 0, stream>>>((float*)d_out, (long long)out_size);
    readout_kernel<<<nblocks, 256, 0, stream>>>(x0, x1, x2, z, batch,
                                                Wo1, bo1, Wo2, bo2, sbias,
                                                (float*)d_out, N);
}
